// DyResBlock_30760555774169
// MI455X (gfx1250) — compile-verified, matrix operand order checked by reading, not run on hardware
//
#include <hip/hip_runtime.h>
#include <math.h>

// ---------------- types ----------------
typedef __attribute__((ext_vector_type(16))) __bf16 v16bf;
typedef __attribute__((ext_vector_type(8)))  float  v8f;
typedef __attribute__((ext_vector_type(4)))  unsigned v4u;
typedef __attribute__((ext_vector_type(8)))  int      v8i;
typedef __attribute__((ext_vector_type(4)))  int      v4i;

struct __attribute__((aligned(16))) U4 { unsigned x, y, z, w; };
struct __attribute__((aligned(16))) U8 { U4 a, b; };

// ---------------- problem constants ----------------
#define BATCH   16
#define CH      192      // C_IN == C_OUT
#define HH      128
#define WW      128
#define KEXP    8
#define LAT     64
#define KTOT    (CH * 9)        // 1728, GEMM K
#define KCHUNK  32
#define NCHUNKS (KTOT / KCHUNK) // 54
#define PITCH   40              // LDS row pitch in uint16 (80 bytes, 16B aligned)
#define HWSZ    (HH * WW)       // 16384

// round-to-nearest-even-ish fp32 -> bf16 (bits)
__device__ __forceinline__ unsigned short f2bf(float f) {
    unsigned u = __builtin_bit_cast(unsigned, f);
    u += 0x7FFFu + ((u >> 16) & 1u);
    return (unsigned short)(u >> 16);
}
__device__ __forceinline__ unsigned pk_bf16(float a, float b) {
    unsigned ua = __builtin_bit_cast(unsigned, a);
    unsigned ub = __builtin_bit_cast(unsigned, b);
    ua += 0x7FFFu + ((ua >> 16) & 1u);
    ub += 0x7FFFu + ((ub >> 16) & 1u);
    return (ua >> 16) | (ub & 0xFFFF0000u);
}
__device__ __forceinline__ float silu(float v) { return v / (1.0f + expf(-v)); }

// Issue a TDM 2D tile load: 64 rows x 32 bf16, row stride KTOT elems, into LDS
// with 80B row pitch (pad 4 DWORDs every 16 DWORDs). ISA cdna5 ch.8 D# layout.
// 6-arg builtin form (clang-23 / therock headers lane).
__device__ __forceinline__ void tdm_load_a(const unsigned short* gsrc, unsigned lds_off) {
    unsigned long long ga = (unsigned long long)(uintptr_t)gsrc;
    v4u g0;
    g0[0] = 1u;                                   // count=1, user descriptor
    g0[1] = lds_off;                              // lds_addr (bytes)
    g0[2] = (unsigned)ga;                         // global_addr[31:0]
    g0[3] = (unsigned)(ga >> 32) | 0x80000000u;   // global_addr[56:32] | type=2<<30
    v8i g1;
    g1[0] = (1 << 16)        // data_size = 2 bytes
          | (1 << 20)        // pad_enable
          | (3 << 22)        // pad_interval: 16 DWORDs (2^(3+1))
          | (3 << 25);       // pad_amount: 4 DWORDs (16 bytes)
    g1[1] = (KCHUNK << 16);  // tensor_dim0 = 32 (bits [79:48] low part)
    g1[2] = (64 << 16);      // tensor_dim1 = 64 (bits [111:80] low part)
    g1[3] = (KCHUNK << 16);  // tile_dim0 = 32 (bits [127:112])
    g1[4] = 64;              // tile_dim1 = 64 (bits [143:128])
    g1[5] = KTOT;            // tensor_dim0_stride = 1728 elems (bits [207:160])
    g1[6] = 0;
    g1[7] = 0;
    v4i z4 = {0, 0, 0, 0};
    v8i z8 = {0, 0, 0, 0, 0, 0, 0, 0};
    __builtin_amdgcn_tensor_load_to_lds(g0, g1, z4, z4, z8, 0);
}

// ---------------- kernel 1: attention softmax [16][8] ----------------
__global__ void attn_kernel(const float* __restrict__ latent,
                            const float* __restrict__ attn_w,
                            const float* __restrict__ attn_b,
                            float* __restrict__ attn) {
    __shared__ float lg[BATCH][KEXP];
    int t = threadIdx.x;                 // 128 threads
    int b = t >> 3, e = t & 7;
    float acc = attn_b[e];
    for (int d = 0; d < LAT; ++d) acc += latent[b * LAT + d] * attn_w[e * LAT + d];
    lg[b][e] = acc;
    __syncthreads();
    float mx = lg[b][0];
    for (int j = 1; j < KEXP; ++j) mx = fmaxf(mx, lg[b][j]);
    float s = 0.f;
    for (int j = 0; j < KEXP; ++j) s += expf(lg[b][j] - mx);
    attn[b * KEXP + e] = expf(lg[b][e] - mx) / s;
}

// ---------------- kernel 2: dynamic bias b_dyn[b][o] ----------------
__global__ void bdyn_kernel(const float* __restrict__ attn,
                            const float* __restrict__ expert_b,
                            float* __restrict__ b_dyn) {
    int b = blockIdx.x, o = threadIdx.x; // 16 blocks x 192 threads
    float s = 0.f;
    for (int e = 0; e < KEXP; ++e) s += attn[b * KEXP + e] * expert_b[e * CH + o];
    b_dyn[b * CH + o] = s;
}

// ---------------- kernel 3: static weights -> bf16, K-fast layout ----------------
// conv2_w[o][i][r] (r = kh*3+kw) -> w2[o][k], k = r*192 + i
__global__ void w2_kernel(const float* __restrict__ conv2_w,
                          unsigned short* __restrict__ w2) {
    __shared__ float buf[KTOT];
    int o = blockIdx.x, t = threadIdx.x; // 192 blocks x 256 threads
    for (int j = t; j < KTOT; j += 256) buf[j] = conv2_w[o * KTOT + j]; // (i*9+r)
    __syncthreads();
    for (int j = t; j < KTOT; j += 256) {
        int r = j / CH, i = j % CH;
        w2[o * KTOT + j] = f2bf(buf[i * 9 + r]);
    }
}

// ---------------- kernel 4: dynamic weights w1[b][o][k] bf16 ----------------
__global__ void w1_kernel(const float* __restrict__ attn,
                          const float* __restrict__ expert_w,
                          unsigned short* __restrict__ w1) {
    __shared__ float buf[KTOT];
    int blk = blockIdx.x;                // 16*192 blocks
    int b = blk / CH, o = blk % CH;
    int t = threadIdx.x;                 // 256 threads
    float rr[7] = {0.f, 0.f, 0.f, 0.f, 0.f, 0.f, 0.f};
    for (int e = 0; e < KEXP; ++e) {
        float a = attn[b * KEXP + e];
        const float* src = expert_w + (size_t)(e * CH + o) * KTOT; // contiguous (i*9+r)
        int jj = 0;
        for (int j = t; j < KTOT; j += 256, ++jj) rr[jj] += a * src[j];
    }
    {
        int jj = 0;
        for (int j = t; j < KTOT; j += 256, ++jj) buf[j] = rr[jj];
    }
    __syncthreads();
    for (int j = t; j < KTOT; j += 256) {
        int r = j / CH, i = j % CH;
        w1[(size_t)(b * CH + o) * KTOT + j] = f2bf(buf[i * 9 + r]);
    }
}

// ---------------- kernel 5: implicit-GEMM 3x3 conv + SiLU (+residual) ----------------
// Block = (b, h, og): M-tile = 64 output channels, N-tile = 128 (full row h).
// K = 1728 as (r=0..8 taps) x (ci=0..5 channel groups of 32).
// A tile staged by the Tensor Data Mover (double buffered); B tile built by all
// threads with clamped loads + select (no divergent branches).
__global__ __launch_bounds__(256) void conv3x3_wmma(
    const float* __restrict__ in,            // [16][192][128][128]
    const unsigned short* __restrict__ wts,  // bf16 [ (b) ][192][1728]
    const float* __restrict__ bias,          // [16][192] or [192]
    const float* __restrict__ resid,         // nullable, same layout as out
    float* __restrict__ out,                 // [16][192][128][128]
    int perBatchW, int perBatchBias) {
    __shared__ __align__(16) unsigned short At[2][64 * PITCH];  // A: 64 x 32 bf16, dbl-buffered
    __shared__ __align__(16) unsigned short Bt[128 * PITCH];    // B: 128 x 32 bf16

    int idx = blockIdx.x;
    int og = idx % 3;
    int h  = (idx / 3) & (HH - 1);
    int b  = idx / (3 * HH);
    int o0 = og * 64;

    int t    = threadIdx.x;
    int lane = t & 31;
    int wv   = t >> 5;
    int mstripe = wv & 3;        // 16-row M stripe within 64
    int nhalf   = wv >> 2;       // 0/1 -> N columns [0,64) or [64,128)
    int khalf   = lane >> 4;     // ISA 16-bit A/B layout: lanes>=16 hold K+8 / K+24
    int lm      = lane & 15;

    // B-tile loader mapping: 16 channels per thread at one spatial column
    int bn = t & 127, bkq = t >> 7;

    const unsigned short* wbase = wts + (perBatchW ? (size_t)b * CH * KTOT : 0)
                                      + (size_t)o0 * KTOT;

    v8f acc[4] = {};

    // prologue: TDM chunk 0 into buffer 0
    if (wv == 0)
        tdm_load_a(wbase, (unsigned)(uintptr_t)&At[0][0]);

    for (int r = 0; r < 9; ++r) {
        // per-tap geometry (hoisted out of the channel-group loop)
        int hh2 = h + (r / 3) - 1;
        int ww2 = bn + (r % 3) - 1;
        bool ok = ((unsigned)hh2 < (unsigned)HH) & ((unsigned)ww2 < (unsigned)WW);
        int hcl = min(max(hh2, 0), HH - 1);
        int wcl = min(max(ww2, 0), WW - 1);
        const float* xrow = in + ((size_t)(b * CH + bkq * 16) * HH + hcl) * WW + wcl;

        for (int ci = 0; ci < 6; ++ci) {
            int c = r * 6 + ci;

            // async-stage A for next chunk into the other buffer
            if (wv == 0 && c + 1 < NCHUNKS)
                tdm_load_a(wbase + (size_t)(c + 1) * KCHUNK,
                           (unsigned)(uintptr_t)&At[(c + 1) & 1][0]);

            // ---- stage B: unconditional clamped loads + select-to-zero
            const float* xsrc = xrow + (size_t)(ci * KCHUNK) * HWSZ;
            unsigned short* bdst = &Bt[bn * PITCH + bkq * 16];
#pragma unroll
            for (int kk = 0; kk < 16; kk += 2) {
                float v0 = xsrc[(size_t)kk * HWSZ];
                float v1 = xsrc[(size_t)(kk + 1) * HWSZ];
                v0 = ok ? v0 : 0.f;
                v1 = ok ? v1 : 0.f;
                *(unsigned*)&bdst[kk] = pk_bf16(v0, v1);
            }
            if (ci + 1 < 6)
                __builtin_prefetch(xsrc + (size_t)KCHUNK * HWSZ, 0, 3);

            // wave0: make sure TDM for chunk c has landed before the barrier
            if (wv == 0) {
                if (c + 1 < NCHUNKS) __builtin_amdgcn_s_wait_tensorcnt(1);
                else                 __builtin_amdgcn_s_wait_tensorcnt(0);
            }
            __syncthreads();

            // ---- fragments per ISA 16-bit layout: two ds_load_b128 each
            const unsigned short* Ab = &At[c & 1][0];
            U8 fa;
            fa.a = *(const U4*)&Ab[(mstripe * 16 + lm) * PITCH + khalf * 8];      // K 0..7 / 8..15
            fa.b = *(const U4*)&Ab[(mstripe * 16 + lm) * PITCH + 16 + khalf * 8]; // K 16..23 / 24..31
            v16bf Am = __builtin_bit_cast(v16bf, fa);
#pragma unroll
            for (int s = 0; s < 4; ++s) {
                int n = nhalf * 64 + s * 16 + lm;
                U8 fb;
                fb.a = *(const U4*)&Bt[n * PITCH + khalf * 8];
                fb.b = *(const U4*)&Bt[n * PITCH + 16 + khalf * 8];
                v16bf Bm = __builtin_bit_cast(v16bf, fb);
                acc[s] = __builtin_amdgcn_wmma_f32_16x16x32_bf16(
                    false, Am, false, Bm, (short)0, acc[s], false, false);
            }
            __syncthreads();
        }
    }

    // ---- epilogue: bias + SiLU (+residual), f32 C/D layout: n=lane&15, m=v+8*(lane>=16)
    const float* bptr = bias + (perBatchBias ? b * CH : 0);
    int mbase = o0 + mstripe * 16 + (lane >> 4) * 8;
#pragma unroll
    for (int s = 0; s < 4; ++s) {
        int nabs = nhalf * 64 + s * 16 + lm;
#pragma unroll
        for (int v = 0; v < 8; ++v) {
            int m = mbase + v;
            float val = acc[s][v] + bptr[m];
            val = silu(val);
            size_t off = ((size_t)(b * CH + m) * HH + h) * WW + nabs;
            if (resid) val += resid[off];
            out[off] = val;
        }
    }
}

// ---------------- launcher ----------------
extern "C" void kernel_launch(void* const* d_in, const int* in_sizes, int n_in,
                              void* d_out, int out_size, void* d_ws, size_t ws_size,
                              hipStream_t stream) {
    const float* x        = (const float*)d_in[0];
    const float* latent   = (const float*)d_in[1];
    const float* expert_w = (const float*)d_in[2];
    const float* expert_b = (const float*)d_in[3];
    const float* attn_w   = (const float*)d_in[4];
    const float* attn_b   = (const float*)d_in[5];
    const float* conv2_w  = (const float*)d_in[6];
    const float* conv2_b  = (const float*)d_in[7];
    float* out = (float*)d_out;

    // workspace carve-up (256B aligned)
    char* ws = (char*)d_ws;
    float*          attn  = (float*)(ws + 0);                       // 16*8*4      = 512
    float*          b_dyn = (float*)(ws + 512);                     // 16*192*4    = 12288
    unsigned short* w2    = (unsigned short*)(ws + 16384);          // 192*1728*2  = 663552
    unsigned short* w1    = (unsigned short*)(ws + 679936);         // 16*192*1728*2 = 10616832
    float*          y1    = (float*)(ws + 11296768);                // 16*192*128*128*4 = 201326592

    attn_kernel<<<1, 128, 0, stream>>>(latent, attn_w, attn_b, attn);
    bdyn_kernel<<<BATCH, CH, 0, stream>>>(attn, expert_b, b_dyn);
    w2_kernel<<<CH, 256, 0, stream>>>(conv2_w, w2);
    w1_kernel<<<BATCH * CH, 256, 0, stream>>>(attn, expert_w, w1);

    int convBlocks = BATCH * HH * 3;  // 6144
    // conv1: dynamic weights + dynamic bias, SiLU, -> y1
    conv3x3_wmma<<<convBlocks, 256, 0, stream>>>(x, w1, b_dyn, nullptr, y1, 1, 1);
    // conv2: static weights + bias, SiLU, +residual x, -> out
    conv3x3_wmma<<<convBlocks, 256, 0, stream>>>(y1, w2, conv2_b, x, out, 0, 0);
}
